// MemoryModule_3100966388141
// MI455X (gfx1250) — matrix operand order repaired and checked by reading, not compiled, not run on hardware
//
#include <hip/hip_runtime.h>

// MI455X / gfx1250 fused memory-attention:
//   attn = softmax(qk^T · mk); out[:,Cv:] = attn · mv^T ; out[:,:Cv] = qv
// BF16 WMMA (v_wmma_f32_16x16x32_bf16), f32 accumulate, flash-style streaming.
// Wave tile: 16 query rows x 256 value channels (16 f32 accumulator tiles).

typedef __attribute__((ext_vector_type(16))) __bf16        v16bf;
typedef __attribute__((ext_vector_type(8)))  float         v8f;
typedef __attribute__((ext_vector_type(8)))  unsigned int  v8u;

#define TT  8
#define BB  8
#define CK  128
#define CV  512
#define HWD 1024
#define THW (TT*HWD)
#define CVT 16        // cv tiles per wave (16 -> 256 channels)
#define PFD 2         // prefetch distance in m-chunks

__device__ __forceinline__ unsigned short f2bf(float x) {
  unsigned int u = __float_as_uint(x);
  u += 0x7FFFu + ((u >> 16) & 1u);          // round-to-nearest-even
  return (unsigned short)(u >> 16);
}
__device__ __forceinline__ unsigned int pack2bf(float lo, float hi) {
  return (unsigned int)f2bf(lo) | ((unsigned int)f2bf(hi) << 16);
}

// 32 contiguous bytes (16 bf16) -> one 8-dword fragment
__device__ __forceinline__ v16bf load_frag32(const unsigned short* p) {
  uint4 a = *(const uint4*)p;
  uint4 b = *(const uint4*)(p + 8);
  v8u r; r[0]=a.x; r[1]=a.y; r[2]=a.z; r[3]=a.w; r[4]=b.x; r[5]=b.y; r[6]=b.z; r[7]=b.w;
  return __builtin_bit_cast(v16bf, r);
}
// two disjoint 16-byte pieces (A-operand: K groups split by 16)
__device__ __forceinline__ v16bf load_frag_split(const unsigned short* p0,
                                                 const unsigned short* p1) {
  uint4 a = *(const uint4*)p0;
  uint4 b = *(const uint4*)p1;
  v8u r; r[0]=a.x; r[1]=a.y; r[2]=a.z; r[3]=a.w; r[4]=b.x; r[5]=b.y; r[6]=b.z; r[7]=b.w;
  return __builtin_bit_cast(v16bf, r);
}

// ---------------- pre-pass conversions ----------------

// mk (T,B,Ck,HW) f32 -> mkT [B][m=T*HW][Ck] bf16   (A-operand friendly)
__global__ __launch_bounds__(256) void conv_mk_kernel(const float* __restrict__ mk,
                                                      unsigned short* __restrict__ mkT) {
  int idx = blockIdx.x * 256 + threadIdx.x;          // B*THW*CK = 2^23
  int c  = idx & (CK - 1);
  int m  = (idx >> 7) & (THW - 1);
  int b  = idx >> 20;
  int t  = m >> 10, hw = m & (HWD - 1);
  mkT[idx] = f2bf(mk[(((size_t)(t * BB + b) * CK + c) * HWD) + hw]);
}

// qk (B,Ck,HW) f32 -> qkT [B][n][Ck] bf16          (B-operand friendly)
__global__ __launch_bounds__(256) void conv_qk_kernel(const float* __restrict__ qk,
                                                      unsigned short* __restrict__ qkT) {
  int idx = blockIdx.x * 256 + threadIdx.x;          // B*HW*CK = 2^20
  int c = idx & (CK - 1);
  int n = (idx >> 7) & (HWD - 1);
  int b = idx >> 17;
  qkT[idx] = f2bf(qk[((size_t)b * CK + c) * HWD + n]);
}

// mv (T,B,Cv,HW) f32 -> same layout bf16           (B-operand friendly, m contiguous)
__global__ __launch_bounds__(256) void conv_mv_kernel(const float* __restrict__ mv,
                                                      unsigned short* __restrict__ mvB) {
  int idx = blockIdx.x * 256 + threadIdx.x;          // T*B*CV*HW = 2^25
  mvB[idx] = f2bf(mv[idx]);
}

// query_value -> first Cv channels of output
__global__ __launch_bounds__(256) void copy_qv_kernel(const float* __restrict__ qv,
                                                      float* __restrict__ out) {
  int idx = blockIdx.x * 256 + threadIdx.x;          // B*CV*HW = 2^22
  int b = idx >> 19;                                 // CV*HW = 2^19
  int r = idx & ((CV * HWD) - 1);
  out[(size_t)b * (2 * CV * HWD) + r] = qv[idx];
}

// ---------------- fused attention ----------------
// grid = (Cv/256, HW/16, B); one wave32 per block.
// m streamed in chunks of 32: S^T via 8 WMMA (K=c, 4 steps), P packs directly into
// A-layout (no cross-lane transpose), then 16 WMMA for O += P*V.
__global__ __launch_bounds__(32) void attn_kernel(
    const unsigned short* __restrict__ mkT,   // [B][THW][CK]
    const unsigned short* __restrict__ qkT,   // [B][HW][CK]
    const unsigned short* __restrict__ mvB,   // [T][B][CV][HW]
    float* __restrict__ out) {                // [B][2*CV][HW]
  const int lane   = threadIdx.x;
  const int lane16 = lane & 15;
  const int half   = lane >> 4;
  const int cv0    = blockIdx.x * (CVT * 16);
  const int n0     = blockIdx.y * 16;
  const int b      = blockIdx.z;

  // Q fragments (B operand): lane -> n = n0+lane16, c = cc*32 + 8*half + [0..7], cc*32 + 16 + 8*half + [0..7]
  v16bf qf[4];
  {
    const unsigned short* qrow =
        qkT + ((size_t)b * HWD + (n0 + lane16)) * CK + half * 8;
#pragma unroll
    for (int cc = 0; cc < 4; ++cc) qf[cc] = load_frag_split(qrow + cc * 32, qrow + cc * 32 + 16);
  }

  v8f o[CVT];
#pragma unroll
  for (int t = 0; t < CVT; ++t) o[t] = v8f{};
  float run_max = -3.0e38f;
  float run_sum = 0.0f;

  const unsigned short* mkb = mkT + (size_t)b * THW * CK;

  for (int chunk = 0; chunk < THW / 32; ++chunk) {
    const int m0 = chunk * 32;

    // ---- S^T tiles: A = mk (M=m, K=c), B = q (K=c, N=n) ----
    v8f s0 = v8f{}, s1 = v8f{};
    const unsigned short* arow0 = mkb + (size_t)(m0 + lane16) * CK + half * 8;
    const unsigned short* arow1 = arow0 + 16 * CK;
    // prefetch the A rows PFD chunks ahead (speculative, counter-free)
    __builtin_prefetch(arow0 + PFD * 32 * CK, 0, 3);
    __builtin_prefetch(arow1 + PFD * 32 * CK, 0, 3);
#pragma unroll
    for (int cc = 0; cc < 4; ++cc) {
      v16bf a0 = load_frag_split(arow0 + cc * 32, arow0 + cc * 32 + 16);
      v16bf a1 = load_frag_split(arow1 + cc * 32, arow1 + cc * 32 + 16);
      s0 = __builtin_amdgcn_wmma_f32_16x16x32_bf16(false, a0, false, qf[cc],
                                                   (short)0, s0, false, false);
      s1 = __builtin_amdgcn_wmma_f32_16x16x32_bf16(false, a1, false, qf[cc],
                                                   (short)0, s1, false, false);
    }

    // ---- online softmax over this 32-wide m chunk ----
    // lane holds S^T[m = v+8*half (+16 for s1)][n = lane16]
    float mloc = s0[0];
#pragma unroll
    for (int v = 1; v < 8; ++v) mloc = fmaxf(mloc, s0[v]);
#pragma unroll
    for (int v = 0; v < 8; ++v) mloc = fmaxf(mloc, s1[v]);
    float mchunk  = fmaxf(mloc, __shfl_xor(mloc, 16, 32));
    float new_max = fmaxf(run_max, mchunk);
    float corr    = __expf(run_max - new_max);
    run_max = new_max;

    float p0[8], p1[8];
    float sloc = 0.0f;
#pragma unroll
    for (int v = 0; v < 8; ++v) {
      p0[v] = __expf(s0[v] - run_max);
      p1[v] = __expf(s1[v] - run_max);
      sloc += p0[v] + p1[v];
    }
    run_sum = run_sum * corr + (sloc + __shfl_xor(sloc, 16, 32));

    // rescale accumulators: O-tile lane holds rows n_local = v + 8*half
#pragma unroll
    for (int v = 0; v < 8; ++v) {
      float rc = __shfl(corr, v + half * 8, 32);
#pragma unroll
      for (int t = 0; t < CVT; ++t) o[t][v] *= rc;
    }

    // ---- P fragment: S^T D-layout maps 1:1 onto second-gemm A-layout ----
    v8u pr;
#pragma unroll
    for (int j = 0; j < 4; ++j) {
      pr[j]     = pack2bf(p0[2 * j], p0[2 * j + 1]);   // K = m {0..7|8..15}
      pr[4 + j] = pack2bf(p1[2 * j], p1[2 * j + 1]);   // K = m {16..23|24..31}
    }
    v16bf pf = __builtin_bit_cast(v16bf, pr);

    // ---- O += P x V : B operand lane -> cv = cv0+t*16+lane16, m = 8*half+[0..7], 16+8*half+[0..7]
    const int tt  = m0 >> 10;
    const int hw0 = m0 & (HWD - 1);
    const unsigned short* vbase =
        mvB + ((size_t)(tt * BB + b) * CV) * HWD + hw0 + half * 8;
#pragma unroll
    for (int t = 0; t < CVT; ++t) {
      const unsigned short* vp = vbase + (size_t)(cv0 + t * 16 + lane16) * HWD;
      v16bf vf = load_frag_split(vp, vp + 16);
      o[t] = __builtin_amdgcn_wmma_f32_16x16x32_bf16(false, pf, false, vf,
                                                     (short)0, o[t], false, false);
    }
  }

  // ---- normalize and store mem -> out[b][Cv + cv][n] ----
  float* ob = out + (size_t)b * (2 * CV * HWD) + (size_t)CV * HWD;
#pragma unroll
  for (int v = 0; v < 8; ++v) {
    float rs  = __shfl(run_sum, v + half * 8, 32);
    float inv = 1.0f / rs;
    int n_abs = n0 + v + half * 8;
#pragma unroll
    for (int t = 0; t < CVT; ++t) {
      int cv_abs = cv0 + t * 16 + lane16;
      ob[(size_t)cv_abs * HWD + n_abs] = o[t][v] * inv;
    }
  }
}

extern "C" void kernel_launch(void* const* d_in, const int* in_sizes, int n_in,
                              void* d_out, int out_size, void* d_ws, size_t ws_size,
                              hipStream_t stream) {
  const float* mk = (const float*)d_in[0];   // (T,B,Ck,H,W)
  const float* mv = (const float*)d_in[1];   // (T,B,Cv,H,W)
  const float* qk = (const float*)d_in[2];   // (B,Ck,H,W)
  const float* qv = (const float*)d_in[3];   // (B,Cv,H,W)
  float* out = (float*)d_out;                // (B,2*Cv,H,W)

  // workspace layout (bf16 staging)
  unsigned char* ws = (unsigned char*)d_ws;
  unsigned short* mkT = (unsigned short*)(ws);                       // 16 MiB
  unsigned short* qkT = (unsigned short*)(ws + (size_t)16777216);    //  2 MiB
  unsigned short* mvB = (unsigned short*)(ws + (size_t)18874368);    // 64 MiB

  conv_mk_kernel<<<(BB * THW * CK) / 256, 256, 0, stream>>>(mk, mkT);
  conv_qk_kernel<<<(BB * HWD * CK) / 256, 256, 0, stream>>>(qk, qkT);
  conv_mv_kernel<<<(TT * BB * CV * HWD) / 256, 256, 0, stream>>>(mv, mvB);
  copy_qv_kernel<<<(BB * CV * HWD) / 256, 256, 0, stream>>>(qv, out);

  dim3 grid(CV / (CVT * 16), HWD / 16, BB);  // (2, 64, 8) wave32 blocks
  attn_kernel<<<grid, 32, 0, stream>>>(mkT, qkT, mvB, out);
}
